// CircleLoss_47794396070460
// MI455X (gfx1250) — matrix operand order repaired and checked by reading, not compiled, not run on hardware
//
#include <hip/hip_runtime.h>
#include <math.h>

// ---------------------------------------------------------------------------
// Circle loss, fused:  normalize -> tiled f16 WMMA Gram matrix -> masked
// relu*exp epilogue -> global sum -> log1p.   sim is never materialized.
// Symmetry: only block-tiles with bj >= bi are computed; off-diagonal tiles
// contribute with weight 2.
// CDNA5 path: global_load_async_to_lds_b128 double-buffered K-chunks,
// s_wait_asynccnt pipelining, v_wmma_f32_16x16x32_f16 accumulation.
// ---------------------------------------------------------------------------

typedef _Float16 v16h __attribute__((ext_vector_type(16)));
typedef _Float16 v8h  __attribute__((ext_vector_type(8)));
typedef float    v8f  __attribute__((ext_vector_type(8)));

#define B_N   8192
#define D_K   256
#define TILE  128
#define KCH   64                  // K-chunk resident in LDS
#define LDSS  (KCH + 8)           // padded row stride (f16): 144B, 16B-aligned
#define CHF16 (TILE * LDSS)       // f16 per chunk buffer per matrix
#define NCH   (D_K / KCH)         // 4 chunks
#define SMEM_BYTES (4 * CHF16 * 2)  // 2 matrices x 2 buffers, bytes

#define GAMMA   2.0f
#define DELTA_P 0.75f             // 1 - margin
#define DELTA_N 0.25f             // margin
#define O_P     1.25f             // 1 + margin
#define O_N    (-0.25f)           // -margin

// ---- CDNA5 async copy helpers --------------------------------------------
__device__ __forceinline__ void async_b128(const _Float16* lds_ptr,
                                           const _Float16* gptr) {
    // LDS byte offset = low 32 bits of the LDS-aperture flat address.
    unsigned lds_off = (unsigned)(unsigned long long)(uintptr_t)lds_ptr;
    asm volatile("global_load_async_to_lds_b128 %0, %1, off"
                 :: "v"(lds_off), "v"(gptr) : "memory");
}

template <int N>
__device__ __forceinline__ void wait_asynccnt() {
    asm volatile("s_wait_asynccnt %0" :: "i"(N) : "memory");
}

__global__ void init_ws(float* ws) {
    ws[0] = 0.0f;
    ws[1] = 0.0f;
}

// One wave (32 lanes) per row: lane handles 8 contiguous floats.
__global__ __launch_bounds__(256) void normalize_rows(
        const float* __restrict__ emb, _Float16* __restrict__ out) {
    const int row  = blockIdx.x * 8 + (threadIdx.x >> 5);
    const int lane = threadIdx.x & 31;
    const float4* p = (const float4*)(emb + (size_t)row * D_K + lane * 8);
    float4 a = p[0];
    float4 b = p[1];
    float ss = a.x*a.x + a.y*a.y + a.z*a.z + a.w*a.w
             + b.x*b.x + b.y*b.y + b.z*b.z + b.w*b.w;
    #pragma unroll
    for (int off = 16; off > 0; off >>= 1) ss += __shfl_xor(ss, off, 32);
    const float r = rsqrtf(ss);
    v8h h;
    h[0] = (_Float16)(a.x * r); h[1] = (_Float16)(a.y * r);
    h[2] = (_Float16)(a.z * r); h[3] = (_Float16)(a.w * r);
    h[4] = (_Float16)(b.x * r); h[5] = (_Float16)(b.y * r);
    h[6] = (_Float16)(b.z * r); h[7] = (_Float16)(b.w * r);
    *(v8h*)(out + (size_t)row * D_K + lane * 8) = h;
}

// Issue one chunk's async copies (4 x b128 per thread, constant count).
__device__ __forceinline__ void stage_chunk(const _Float16* __restrict__ N16,
                                            int i0, int j0, int kc,
                                            _Float16* ldsA, _Float16* ldsB,
                                            int t) {
    #pragma unroll
    for (int u = 0; u < 2; ++u) {
        const int c   = t * 2 + u;          // 0..511
        const int row = c >> 2;
        const int seg = c & 3;
        const size_t ga = (size_t)(i0 + row) * D_K + kc * KCH + seg * 8;
        const size_t gb = (size_t)(j0 + row) * D_K + kc * KCH + seg * 8;
        async_b128(ldsA + row * LDSS + seg * 8, N16 + ga);
        async_b128(ldsB + row * LDSS + seg * 8, N16 + gb);
    }
}

// 128x128 similarity tile per block (8 waves, wave32).  Wave w owns a 32x64
// region = 2x4 subtiles of 16x16, accumulated with v_wmma_f32_16x16x32_f16.
__global__ __launch_bounds__(256) void circle_tiles(
        const _Float16* __restrict__ N16, const int* __restrict__ labels,
        float* __restrict__ ws) {
    const int bi = blockIdx.x;
    const int bj = blockIdx.y;
    if (bj < bi) return;                    // block-uniform: upper triangle
    const int   i0  = bi * TILE;
    const int   j0  = bj * TILE;
    const float wgt = (bi == bj) ? 1.0f : 2.0f;

    extern __shared__ __align__(16) _Float16 smem[];
    _Float16* ldsA = smem;                  // [2 buf][TILE][LDSS]
    _Float16* ldsB = smem + 2 * CHF16;      // [2 buf][TILE][LDSS]

    __shared__ int   labI[TILE];
    __shared__ int   labJ[TILE];
    __shared__ float red[16];

    const int t = threadIdx.x;
    if (t < TILE) labI[t] = labels[i0 + t];
    else          labJ[t - TILE] = labels[j0 + (t - TILE)];

    const int lane = t & 31;
    const int wv   = t >> 5;
    const int wm   = wv >> 1;               // 0..3 : 32-row group
    const int wn   = wv & 1;                // 0..1 : 64-col group

    v8f zero = {};
    v8f acc[2][4];
    #pragma unroll
    for (int sm = 0; sm < 2; ++sm)
        #pragma unroll
        for (int sn = 0; sn < 4; ++sn) acc[sm][sn] = zero;

    const int mr = lane & 15;               // A row / B col within subtile
    const int ka = (lane < 16) ? 0 : 8;     // A-matrix lane K base
    const int kb = (lane < 16) ? 0 : 8;     // B-matrix lane K base

    // Prologue: prefetch chunk 0 into buffer 0.
    stage_chunk(N16, i0, j0, 0, ldsA, ldsB, t);

    #pragma unroll
    for (int kc = 0; kc < NCH; ++kc) {
        const int cur = kc & 1;
        // Prefetch next chunk into the other buffer, then wait until only
        // those 4 just-issued ops remain outstanding (async ops complete
        // in order => current chunk is resident in LDS).
        if (kc + 1 < NCH) {
            stage_chunk(N16, i0, j0, kc + 1,
                        ldsA + ((kc + 1) & 1) * CHF16,
                        ldsB + ((kc + 1) & 1) * CHF16, t);
            wait_asynccnt<4>();
        } else {
            wait_asynccnt<0>();
        }
        __syncthreads();

        const _Float16* Abuf = ldsA + cur * CHF16;
        const _Float16* Bbuf = ldsB + cur * CHF16;

        #pragma unroll
        for (int ks = 0; ks < KCH; ks += 32) {
            v16h afrag[2];
            v16h bfrag[4];
            #pragma unroll
            for (int sm = 0; sm < 2; ++sm) {
                const _Float16* pa =
                    Abuf + (wm * 32 + sm * 16 + mr) * LDSS + ks + ka;
                *((v8h*)&afrag[sm])       = *(const v8h*)pa;        // K 0..7
                *(((v8h*)&afrag[sm]) + 1) = *(const v8h*)(pa + 16); // K 16..23
            }
            #pragma unroll
            for (int sn = 0; sn < 4; ++sn) {
                const _Float16* pb =
                    Bbuf + (wn * 64 + sn * 16 + mr) * LDSS + ks + kb;
                *((v8h*)&bfrag[sn])       = *(const v8h*)pb;        // K 0..7
                *(((v8h*)&bfrag[sn]) + 1) = *(const v8h*)(pb + 16); // K 16..23
            }
            #pragma unroll
            for (int sm = 0; sm < 2; ++sm)
                #pragma unroll
                for (int sn = 0; sn < 4; ++sn)
                    acc[sm][sn] = __builtin_amdgcn_wmma_f32_16x16x32_f16(
                        false, afrag[sm], false, bfrag[sn],
                        (short)0, acc[sm][sn], false, false);
        }
        // All waves done reading this buffer before the next prefetch
        // (issued at top of next iteration) may overwrite it.
        __syncthreads();
    }

    // Epilogue: decode C layout (VGPR r -> M=r / r+8 by lane half, N=lane&15)
    float pos = 0.0f, neg = 0.0f;
    const int mbase = (lane < 16) ? 0 : 8;
    #pragma unroll
    for (int sm = 0; sm < 2; ++sm) {
        #pragma unroll
        for (int sn = 0; sn < 4; ++sn) {
            const int lj = labJ[wn * 64 + sn * 16 + (lane & 15)];
            #pragma unroll
            for (int r = 0; r < 8; ++r) {
                const int   li = labI[wm * 32 + sm * 16 + r + mbase];
                const float s  = acc[sm][sn][r];
                if (li == lj) {
                    const float ap = fmaxf(s - DELTA_P, 0.0f);
                    pos += ap * __expf(-GAMMA * (s - O_P));
                } else {
                    const float an = fmaxf(DELTA_N - s, 0.0f);
                    neg += an * __expf(GAMMA * (s - O_N));
                }
            }
        }
    }
    pos *= wgt;
    neg *= wgt;
    #pragma unroll
    for (int off = 16; off > 0; off >>= 1) {
        pos += __shfl_xor(pos, off, 32);
        neg += __shfl_xor(neg, off, 32);
    }
    if (lane == 0) { red[wv] = pos; red[8 + wv] = neg; }
    __syncthreads();
    if (t == 0) {
        float P = 0.0f, Q = 0.0f;
        #pragma unroll
        for (int i = 0; i < 8; ++i) { P += red[i]; Q += red[8 + i]; }
        atomicAdd(&ws[0], P);
        atomicAdd(&ws[1], Q);
    }
}

__global__ void finalize(const float* __restrict__ ws, float* __restrict__ out) {
    out[0] = logf(1.0f + ws[0] + ws[1]);
}

extern "C" void kernel_launch(void* const* d_in, const int* in_sizes, int n_in,
                              void* d_out, int out_size, void* d_ws, size_t ws_size,
                              hipStream_t stream) {
    const float* emb    = (const float*)d_in[0];
    const int*   labels = (const int*)d_in[1];
    float*       out    = (float*)d_out;
    float*       wsf    = (float*)d_ws;
    _Float16*    N16    = (_Float16*)((char*)d_ws + 256);   // 4 MB f16 matrix

    init_ws<<<1, 1, 0, stream>>>(wsf);
    normalize_rows<<<B_N / 8, 256, 0, stream>>>(emb, N16);
    circle_tiles<<<dim3(B_N / TILE, B_N / TILE), 256, SMEM_BYTES, stream>>>(
        N16, labels, wsf);
    finalize<<<1, 1, 0, stream>>>(wsf, out);
}
